// Net_20701742367009
// MI455X (gfx1250) — compile-verified, matrix operand order checked by reading, not run on hardware
//
#include <hip/hip_runtime.h>
#include <hip/hip_bf16.h>
#include <math.h>

typedef __attribute__((ext_vector_type(16))) _Float16 v16h;
typedef __attribute__((ext_vector_type(8)))  _Float16 v8h;
typedef __attribute__((ext_vector_type(8)))  float    v8f;

#define NB   256   // graphs
#define NP   512   // points per graph
#define NH   16    // hidden dim
#define NK   8     // neighbors
#define NIN  4

union HFrag { v16h f; v8h h[2]; };

__device__ __forceinline__ float elu1(float x) {
    return x > 0.f ? x : (__expf(x) - 1.f);
}

// Wave-local LDS RAW fence: LDS ops of a wave complete in order; this waits
// DScnt==0 and stops the compiler from reordering LDS accesses across it.
__device__ __forceinline__ void lds_wave_fence() {
    asm volatile("s_wait_dscnt 0" ::: "memory");
}

// ---------------------------------------------------------------------------
// Kernel 1: encoder  h = elu(elu(x @ We1 + be1) @ We2 + be2)
// ---------------------------------------------------------------------------
__global__ __launch_bounds__(256) void encoder_kernel(
    const float* __restrict__ x,
    const float* __restrict__ W1, const float* __restrict__ b1,
    const float* __restrict__ W2, const float* __restrict__ b2,
    float* __restrict__ h)
{
    int i = blockIdx.x * blockDim.x + threadIdx.x;
    if (i >= NB * NP) return;

    float4 xin = ((const float4*)x)[i];          // IN_DIM == 4
    float xv[NIN] = {xin.x, xin.y, xin.z, xin.w};

    float h1[NH];
    #pragma unroll
    for (int j = 0; j < NH; ++j) {
        float s = b1[j];
        #pragma unroll
        for (int d = 0; d < NIN; ++d) s += xv[d] * W1[d * NH + j];
        h1[j] = elu1(s);
    }
    float h2[NH];
    #pragma unroll
    for (int j = 0; j < NH; ++j) {
        float s = b2[j];
        #pragma unroll
        for (int d = 0; d < NH; ++d) s += h1[d] * W2[d * NH + j];
        h2[j] = elu1(s);
    }
    float4* dst = (float4*)(h + (size_t)i * NH);
    #pragma unroll
    for (int q = 0; q < 4; ++q)
        dst[q] = make_float4(h2[4*q], h2[4*q+1], h2[4*q+2], h2[4*q+3]);
}

// ---------------------------------------------------------------------------
// Kernel 2: fused EdgeConv. One workgroup (128 thr = 4 waves) per graph.
// Waves are fully decoupled after stage 0 (per-wave LDS scratch, no block
// barriers in the hot loop — only wave-local s_wait_dscnt fences).
// ---------------------------------------------------------------------------
__global__ __launch_bounds__(128) void edgeconv_kernel(
    const float* __restrict__ fin,
    const float* __restrict__ Wc, const float* __restrict__ bc,
    float* __restrict__ fout)
{
    __shared__ __align__(16) _Float16 sh[NP * NH];      // 16 KB  features f16
    __shared__ __align__(16) float    ssq[NP];          //  2 KB  sq norms
    __shared__ __align__(16) _Float16 sW[2 * NH * NH];  //  1 KB  W_c f16
    __shared__ __align__(16) float    sd2t[4][16 * 16]; //  4 KB  per-wave d2
    __shared__ __align__(16) int      snbr[4][16 * NK]; //  2 KB  per-wave nbrs

    const int g     = blockIdx.x;
    const int tid   = threadIdx.x;
    const int wave  = tid >> 5;
    const int lane  = tid & 31;
    const int lhalf = lane >> 4;
    const int l15   = lane & 15;

    const float* F = fin + (size_t)g * NP * NH;

    // Stage 0: features -> f16 LDS (b128 global loads), squared norms, W_c.
    for (int r = tid; r < NP; r += 128) {
        const float4* src = (const float4*)(F + r * NH);
        float4 q0 = src[0], q1 = src[1], q2 = src[2], q3 = src[3];
        float v[16] = {q0.x,q0.y,q0.z,q0.w, q1.x,q1.y,q1.z,q1.w,
                       q2.x,q2.y,q2.z,q2.w, q3.x,q3.y,q3.z,q3.w};
        float sq = 0.f;
        v8h p0, p1;
        #pragma unroll
        for (int c = 0; c < 8; ++c) {
            sq = fmaf(v[c], v[c], sq);
            sq = fmaf(v[c+8], v[c+8], sq);
            p0[c] = (_Float16)v[c];
            p1[c] = (_Float16)v[c+8];
        }
        v8h* dst = (v8h*)(sh + r * NH);
        dst[0] = p0; dst[1] = p1;
        ssq[r] = sq;
    }
    for (int i = tid; i < 2 * NH * NH; i += 128) sW[i] = (_Float16)Wc[i];
    __syncthreads();

    const float bias = bc[l15];          // this lane's output channel bias
    const v8h zero8 = {};

    // W_c B-fragment (32x16, KxN): invariant; strided gather (built once).
    HFrag wfrag;
    #pragma unroll
    for (int v = 0; v < 8; ++v) {
        int k = (v < 4 ? 2 * v : 16 + 2 * (v - 4)) + 8 * lhalf;
        wfrag.f[2 * v]     = sW[k * NH + l15];
        wfrag.f[2 * v + 1] = sW[(k + 1) * NH + l15];
    }

    float* myd2  = &sd2t[wave][0];
    int*   mynbr = &snbr[wave][0];

    // Row tiles of 16 points; waves work on independent tiles.
    for (int rt = wave; rt < NP / 16; rt += 4) {
        const int m0 = rt * 16;

        // A-fragment: one b128 LDS load (live K chunk), upper half = K-pad.
        HFrag afrag;
        afrag.h[0] = *((const v8h*)(sh + (m0 + l15) * NH + 8 * lhalf));
        afrag.h[1] = zero8;

        float bd[NK]; int bi[NK];
        #pragma unroll
        for (int j = 0; j < NK; ++j) { bd[j] = 3.4e38f; bi[j] = 0; }

        for (int ct = 0; ct < NP / 16; ++ct) {
            const int n0 = ct * 16;
            HFrag bfrag;                   // B = h^T tile: same chunk, N-lane
            bfrag.h[0] = *((const v8h*)(sh + (n0 + l15) * NH + 8 * lhalf));
            bfrag.h[1] = zero8;

            v8f cfr = {};
            cfr = __builtin_amdgcn_wmma_f32_16x16x32_f16(
                false, afrag.f, false, bfrag.f, (short)0, cfr, false, false);

            // d2 = |hi|^2 + |hj|^2 - 2 hi.hj ; D element (r+8*half, l15)
            #pragma unroll
            for (int r = 0; r < 8; ++r) {
                int Ml = r + 8 * lhalf;
                myd2[Ml * 16 + l15] =
                    ssq[m0 + Ml] + ssq[n0 + l15] - 2.0f * cfr[r];
            }
            lds_wave_fence();
            if (lane < 16) {               // lane owns row `lane`
                const float4* rowq = (const float4*)(myd2 + lane * 16);
                #pragma unroll
                for (int qi = 0; qi < 4; ++qi) {
                    float4 q = rowq[qi];
                    float cand[4] = {q.x, q.y, q.z, q.w};
                    #pragma unroll
                    for (int c = 0; c < 4; ++c) {
                        float d = cand[c];
                        if (d < bd[NK - 1]) {
                            bd[NK - 1] = d; bi[NK - 1] = n0 + 4 * qi + c;
                            #pragma unroll
                            for (int s = NK - 1; s > 0; --s) {
                                if (bd[s] < bd[s - 1]) {
                                    float td = bd[s]; bd[s] = bd[s-1]; bd[s-1] = td;
                                    int   ti = bi[s]; bi[s] = bi[s-1]; bi[s-1] = ti;
                                }
                            }
                        }
                    }
                }
            }
            lds_wave_fence();
        }

        if (lane < 16) {
            int4* nq = (int4*)(mynbr + lane * NK);
            nq[0] = make_int4(bi[0], bi[1], bi[2], bi[3]);
            nq[1] = make_int4(bi[4], bi[5], bi[6], bi[7]);
        }
        lds_wave_fence();

        // Edge stage: 16 rows x 8 nbrs = 128 edges = 8 WMMA tiles.
        // Lane's 8 D regs = one point's 8 edge values for channel l15.
        for (int et = 0; et < 8; ++et) {
            int ge  = et * 16 + l15;
            int row = ge >> 3;
            int kk  = ge & 7;
            int nb  = mynbr[row * NK + kk];
            v8h xi = *((const v8h*)(sh + (m0 + row) * NH + 8 * lhalf));
            v8h xj = *((const v8h*)(sh + nb * NH + 8 * lhalf));
            HFrag efrag;                    // e = [xi, xj - xi], K=32 exact
            efrag.h[0] = xi;
            efrag.h[1] = xj - xi;           // v_pk_sub_f16

            v8f mfr = {};
            mfr = __builtin_amdgcn_wmma_f32_16x16x32_f16(
                false, efrag.f, false, wfrag.f, (short)0, mfr, false, false);

            float mx = -3.4e38f;
            #pragma unroll
            for (int r = 0; r < 8; ++r)
                mx = fmaxf(mx, elu1(mfr[r] + bias));

            int orow = m0 + 2 * et + lhalf;
            fout[((size_t)g * NP + orow) * NH + l15] = mx;
        }
    }
}

// ---------------------------------------------------------------------------
// Kernel 3: mean-pool over 512 points + 16->8->4->1 head; one block per graph.
// ---------------------------------------------------------------------------
__global__ __launch_bounds__(128) void head_kernel(
    const float* __restrict__ f2,
    const float* __restrict__ Wo1, const float* __restrict__ bo1,
    const float* __restrict__ Wo2, const float* __restrict__ bo2,
    const float* __restrict__ Wo3, const float* __restrict__ bo3,
    float* __restrict__ out)
{
    __shared__ float acc[128];
    __shared__ float pooled[NH];
    const int g = blockIdx.x;
    const int t = threadIdx.x;
    const int c = t & 15, rg = t >> 4;   // 8 row-groups x 16 channels

    float s = 0.f;
    for (int r = rg; r < NP; r += 8)
        s += f2[((size_t)g * NP + r) * NH + c];
    acc[t] = s;
    __syncthreads();

    if (t < NH) {
        float tot = 0.f;
        #pragma unroll
        for (int k = 0; k < 8; ++k) tot += acc[k * 16 + t];
        pooled[t] = tot * (1.0f / (float)NP);
    }
    __syncthreads();

    if (t == 0) {
        float o1[8];
        #pragma unroll
        for (int j = 0; j < 8; ++j) {
            float v = bo1[j];
            #pragma unroll
            for (int d = 0; d < NH; ++d) v += pooled[d] * Wo1[d * 8 + j];
            o1[j] = elu1(v);
        }
        float o2[4];
        #pragma unroll
        for (int j = 0; j < 4; ++j) {
            float v = bo2[j];
            #pragma unroll
            for (int d = 0; d < 8; ++d) v += o1[d] * Wo2[d * 4 + j];
            o2[j] = elu1(v);
        }
        float o = bo3[0];
        #pragma unroll
        for (int d = 0; d < 4; ++d) o += o2[d] * Wo3[d];
        out[g] = o;
        ((int*)out)[NB + g] = g;   // pooled_batch (int32), concatenated
    }
}

// ---------------------------------------------------------------------------
extern "C" void kernel_launch(void* const* d_in, const int* in_sizes, int n_in,
                              void* d_out, int out_size, void* d_ws, size_t ws_size,
                              hipStream_t stream) {
    const float* x_pf = (const float*)d_in[0];
    // d_in[1] batch_pf, d_in[2] num_graphs : unused (fixed B=256, P=512)
    const float* W_e1 = (const float*)d_in[3];
    const float* b_e1 = (const float*)d_in[4];
    const float* W_e2 = (const float*)d_in[5];
    const float* b_e2 = (const float*)d_in[6];
    const float* W_c  = (const float*)d_in[7];
    const float* b_c  = (const float*)d_in[8];
    const float* W_o1 = (const float*)d_in[9];
    const float* b_o1 = (const float*)d_in[10];
    const float* W_o2 = (const float*)d_in[11];
    const float* b_o2 = (const float*)d_in[12];
    const float* W_o3 = (const float*)d_in[13];
    const float* b_o3 = (const float*)d_in[14];

    const size_t feat_elems = (size_t)NB * NP * NH;   // 2,097,152 floats
    float* h  = (float*)d_ws;                          // 8 MB
    float* f1 = h + feat_elems;                        // 8 MB

    encoder_kernel<<<(NB * NP) / 256, 256, 0, stream>>>(
        x_pf, W_e1, b_e1, W_e2, b_e2, h);

    edgeconv_kernel<<<NB, 128, 0, stream>>>(h,  W_c, b_c, f1);   // layer 1
    edgeconv_kernel<<<NB, 128, 0, stream>>>(f1, W_c, b_c, h);    // layer 2

    head_kernel<<<NB, 128, 0, stream>>>(
        h, W_o1, b_o1, W_o2, b_o2, W_o3, b_o3, (float*)d_out);
}
